// SceneGraph_11605001634431
// MI455X (gfx1250) — matrix operand order repaired and checked by reading, not compiled, not run on hardware
//
#include <hip/hip_runtime.h>
#include <math.h>

// ---------------- types ----------------
typedef __attribute__((ext_vector_type(16))) __bf16 bf16x16;
typedef __attribute__((ext_vector_type(8)))  __bf16 bf16x8;
typedef __attribute__((ext_vector_type(8)))  float  f32x8;

// ---------------- generic bf16 WMMA GEMM ----------------
// A: [M][K] bf16 row-major, BT: [N][K] bf16 row-major (i.e. B transposed)
// out: [M][N] f32 = A x B + bias[m]
// One wave computes a 16 x (16*NW) tile. K multiple of 32. N multiple of 16*NW.
// Fragment layouts (wave32 CDNA5):
//  A: lane L row m=M0+(L&15); elems j<8 -> K=k0+8*(L>>4)+j ; j>=8 -> K=k0+16+8*(L>>4)+(j-8)
//     -> two contiguous 8-element (16B) runs.
//  B: lane L col n=N0+(L&15); same K mapping as A -> two contiguous 8-element (16B) runs.
//  C/D: lane L: n=N0+(L&15); vgpr v: m=M0+v+8*(L>>4)
template <int NW>
__global__ void k_gemm_bf16(const __bf16* __restrict__ A, const __bf16* __restrict__ BT,
                            const float* __restrict__ bias, float* __restrict__ out,
                            int Mt, int N, int K) {
  const int wave = (blockIdx.x * blockDim.x + threadIdx.x) >> 5;
  const int nGroups = (N >> 4) / NW;
  const int mt = wave % Mt;
  const int ng = wave / Mt;
  if (ng >= nGroups) return;
  const int M0 = mt * 16;
  const int N0 = ng * 16 * NW;
  const int lane = threadIdx.x & 31;
  const int half = lane >> 4;
  const int l15  = lane & 15;

  const __bf16* Arow = A + (size_t)(M0 + l15) * K + 8 * half;
  const __bf16* Brow[NW];
#pragma unroll
  for (int i = 0; i < NW; ++i)
    Brow[i] = BT + (size_t)(N0 + i * 16 + l15) * K + 8 * half;

  f32x8 acc[NW];
#pragma unroll
  for (int i = 0; i < NW; ++i) acc[i] = (f32x8){};

  for (int k0 = 0; k0 < K; k0 += 32) {
    if (k0 + 64 <= K) {
      __builtin_prefetch(Arow + k0 + 32, 0, 2);
#pragma unroll
      for (int i = 0; i < NW; ++i) __builtin_prefetch(Brow[i] + k0 + 32, 0, 2);
    }
    bf16x8 alo = *(const bf16x8*)(Arow + k0);
    bf16x8 ahi = *(const bf16x8*)(Arow + k0 + 16);
    bf16x16 a = __builtin_shufflevector(alo, ahi, 0, 1, 2, 3, 4, 5, 6, 7,
                                        8, 9, 10, 11, 12, 13, 14, 15);
#pragma unroll
    for (int i = 0; i < NW; ++i) {
      bf16x8 blo = *(const bf16x8*)(Brow[i] + k0);
      bf16x8 bhi = *(const bf16x8*)(Brow[i] + k0 + 16);
      bf16x16 b = __builtin_shufflevector(blo, bhi, 0, 1, 2, 3, 4, 5, 6, 7,
                                          8, 9, 10, 11, 12, 13, 14, 15);
      acc[i] = __builtin_amdgcn_wmma_f32_16x16x32_bf16(false, a, false, b, (short)0,
                                                       acc[i], false, false);
    }
  }
#pragma unroll
  for (int i = 0; i < NW; ++i) {
    const int nn = N0 + i * 16 + l15;
#pragma unroll
    for (int v = 0; v < 8; ++v) {
      const int m = M0 + v + 8 * half;
      out[(size_t)m * N + nn] = acc[i][v] + bias[m];
    }
  }
}

// ---------------- small helpers ----------------
__device__ __forceinline__ float hat_int(float t) {
  t = fminf(fmaxf(t, -1.0f), 1.0f);
  return (t < 0.0f) ? 0.5f * (t + 1.0f) * (t + 1.0f)
                    : 1.0f - 0.5f * (1.0f - t) * (1.0f - t);
}

// ---------------- pack / convert kernels ----------------
__global__ void k_cvt_bf16(const float* __restrict__ s, __bf16* __restrict__ d, int n) {
  int t = blockIdx.x * blockDim.x + threadIdx.x;
  if (t < n) d[t] = (__bf16)s[t];
}

// input [8,256,32,32] -> BT1 [8192][256] bf16 (column x = b*1024+hw, row k = channel)
__global__ void k_pack_inT(const float* __restrict__ in, __bf16* __restrict__ bt) {
  int t = blockIdx.x * blockDim.x + threadIdx.x;
  if (t >= 8192 * 256) return;
  int x = t >> 8, c = t & 255;
  bt[t] = (__bf16)in[(size_t)((x >> 10) * 256 + c) * 1024 + (x & 1023)];
}

// obj_in^T [784][512]: k<256 obj_pool, 256..383 x_c, 384..511 y_c*imap
__global__ void k_pack_objinT(const float* __restrict__ objp, const float* __restrict__ ctxp,
                              const float* __restrict__ imap, __bf16* __restrict__ bt) {
  int t = blockIdx.x * blockDim.x + threadIdx.x;
  if (t >= 784 * 512) return;
  int x = t >> 9, k = t & 511;
  unsigned roi = (unsigned)x / 49u, s = (unsigned)x - roi * 49u;
  float v;
  if (k < 256) v = objp[(roi * 256 + k) * 49 + s];
  else {
    v = ctxp[(k - 256) * 49 + s];
    if (k >= 384) v *= imap[roi * 49 + s];
  }
  bt[t] = (__bf16)v;
}

// rel_in^T [12544][896]: objf[sub], objf[obj], masked rel_pool
__global__ void k_pack_relinT(const float* __restrict__ objf, const float* __restrict__ relp,
                              const float* __restrict__ simap, const float* __restrict__ oimap,
                              __bf16* __restrict__ bt) {
  unsigned t = blockIdx.x * blockDim.x + threadIdx.x;
  if (t >= 12544u * 896u) return;
  unsigned x = t / 896u, k = t - x * 896u;
  unsigned pr = x / 49u, s = x - pr * 49u;
  float v;
  if (k < 256) v = objf[k * 784 + (pr >> 4) * 49 + s];
  else if (k < 512) v = objf[(k - 256) * 784 + (pr & 15) * 49 + s];
  else {
    unsigned c = k - 512;
    v = relp[((size_t)pr * 384 + c) * 49 + s];
    if (c >= 256) v *= oimap[pr * 49 + s];
    else if (c >= 128) v *= simap[pr * 49 + s];
  }
  bt[t] = (__bf16)v;
}

// relu(rel_feat)^T [256 pairs][12544] from relf [256 ch][12544]
__global__ void k_pack_relfT(const float* __restrict__ relf, __bf16* __restrict__ bt) {
  unsigned t = blockIdx.x * blockDim.x + threadIdx.x;
  if (t >= 256u * 12544u) return;
  unsigned p = t / 12544u, k = t - p * 12544u;
  unsigned ch = k / 49u, s = k - ch * 49u;
  float v = relf[(size_t)ch * 12544 + p * 49 + s];
  bt[t] = (__bf16)(v > 0.0f ? v : 0.0f);
}

// relu(obj_feat)^T [16][12544] from objf [256 ch][784]
__global__ void k_pack_objfT(const float* __restrict__ objf, __bf16* __restrict__ bt) {
  unsigned t = blockIdx.x * blockDim.x + threadIdx.x;
  if (t >= 16u * 12544u) return;
  unsigned nr = t / 12544u, k = t - nr * 12544u;
  unsigned ch = k / 49u, s = k - ch * 49u;
  float v = objf[ch * 784 + nr * 49 + s];
  bt[t] = (__bf16)(v > 0.0f ? v : 0.0f);
}

// ---------------- setup kernels ----------------
__global__ void k_init_imgbox(float* ib) {
  if (blockIdx.x == 0 && threadIdx.x == 0) { ib[0] = 0.f; ib[1] = 0.f; ib[2] = 512.f; ib[3] = 512.f; }
}

__global__ void k_roi_weights(const float* boxes, int R, float* Wx, float* Wy, float* inv_area) {
  int t = blockIdx.x * blockDim.x + threadIdx.x;
  if (t >= R * 224) return;
  int r = t / 224, rem = t % 224;
  int p = rem / 32, i = rem % 32;
  const float sc = 1.0f / 16.0f;
  float x0 = boxes[r * 4 + 0] * sc, y0 = boxes[r * 4 + 1] * sc;
  float x1 = boxes[r * 4 + 2] * sc, y1 = boxes[r * 4 + 3] * sc;
  float bwx = (x1 - x0) * (1.0f / 7.0f), bwy = (y1 - y0) * (1.0f / 7.0f);
  float fi = (float)i;
  float sx = x0 + bwx * (float)p, ex = sx + bwx;
  Wx[r * 224 + p * 32 + i] = hat_int(ex - fi) - hat_int(sx - fi);
  float sy = y0 + bwy * (float)p, ey = sy + bwy;
  Wy[r * 224 + p * 32 + i] = hat_int(ey - fi) - hat_int(sy - fi);
  if (rem == 0) inv_area[r] = 1.0f / fmaxf(bwx * bwy, 1e-8f);
}

__global__ void k_unions(const float* objects, float* uni) {
  int t = blockIdx.x * blockDim.x + threadIdx.x;
  if (t >= 2048) return;
  int b = t >> 8, pr = t & 255;
  const float* S = objects + (b * 16 + (pr >> 4)) * 4;
  const float* O = objects + (b * 16 + (pr & 15)) * 4;
  uni[t * 4 + 0] = fminf(S[0], O[0]);
  uni[t * 4 + 1] = fminf(S[1], O[1]);
  uni[t * 4 + 2] = fmaxf(S[2], O[2]);
  uni[t * 4 + 3] = fmaxf(S[3], O[3]);
}

__global__ void k_imap_img(const float* objects, float* out) {
  int t = blockIdx.x * blockDim.x + threadIdx.x;
  if (t >= 128 * 49) return;
  int pq = t % 49, rb = t / 49;
  int p = pq / 7, q = pq % 7;
  const float* A = objects + rb * 4;
  const float bw = 512.0f / 7.0f;
  float gx0 = bw * (float)q, gy0 = bw * (float)p;
  float ox = fmaxf(fminf(gx0 + bw, A[2]) - fmaxf(gx0, A[0]), 0.0f);
  float oy = fmaxf(fminf(gy0 + bw, A[3]) - fmaxf(gy0, A[1]), 0.0f);
  out[t] = oy * ox / fmaxf(bw * bw, 1e-8f);
}

__global__ void k_imap_pairs(const float* objects, const float* uni, float* simap, float* oimap) {
  int t = blockIdx.x * blockDim.x + threadIdx.x;
  if (t >= 2048 * 49) return;
  int pq = t % 49, tp = t / 49;
  int p = pq / 7, q = pq % 7;
  int b = tp >> 8, pr = tp & 255;
  const float* U = uni + tp * 4;
  const float* S = objects + (b * 16 + (pr >> 4)) * 4;
  const float* O = objects + (b * 16 + (pr & 15)) * 4;
  float bw = (U[2] - U[0]) * (1.0f / 7.0f), bh = (U[3] - U[1]) * (1.0f / 7.0f);
  float gx0 = U[0] + bw * (float)q, gy0 = U[1] + bh * (float)p;
  float inv = 1.0f / fmaxf(bw * bh, 1e-8f);
  float oxs = fmaxf(fminf(gx0 + bw, S[2]) - fmaxf(gx0, S[0]), 0.0f);
  float oys = fmaxf(fminf(gy0 + bh, S[3]) - fmaxf(gy0, S[1]), 0.0f);
  simap[t] = oys * oxs * inv;
  float oxo = fmaxf(fminf(gx0 + bw, O[2]) - fmaxf(gx0, O[0]), 0.0f);
  float oyo = fmaxf(fminf(gy0 + bh, O[3]) - fmaxf(gy0, O[1]), 0.0f);
  oimap[t] = oyo * oxo * inv;
}

// ---------------- PrRoI pooling ----------------
// feat: channel c at feat + c*cstride, 32x32 row-major. out: [R,C,49]
__global__ void k_prroi_pool(const float* feat, int C, int cstride, const float* Wy,
                             const float* Wx, const float* inv_area, int R, float* out) {
  int t = blockIdx.x * blockDim.x + threadIdx.x;
  if (t >= R * C * 7) return;
  int p = t % 7;
  int rc = t / 7;
  int c = rc % C, r = rc / C;
  const float* wy = Wy + r * 224 + p * 32;
  const float* wx = Wx + r * 224;
  const float* f  = feat + (size_t)c * cstride;
  float col[32];
#pragma unroll
  for (int w = 0; w < 32; ++w) col[w] = 0.0f;
  for (int h = 0; h < 32; ++h) {
    float wv = wy[h];
    if (wv != 0.0f) {
      const float* row = f + h * 32;
#pragma unroll
      for (int w = 0; w < 32; ++w) col[w] += wv * row[w];
    }
  }
  float ia = inv_area[r];
  float* o = out + (size_t)(r * C + c) * 49 + p * 7;
#pragma unroll
  for (int q = 0; q < 7; ++q) {
    const float* wq = wx + q * 32;
    float acc = 0.0f;
#pragma unroll
    for (int w = 0; w < 32; ++w) acc += col[w] * wq[w];
    o[q] = acc * ia;
  }
}

// L2-normalize columns of tmp [256, N]; dst[j*256+d] = tmp[d,j]/||.||
__global__ void k_normalize(const float* tmp, int N, float* dst) {
  __shared__ float red[256];
  int j = blockIdx.x, d = threadIdx.x;
  float v = tmp[d * N + j];
  red[d] = v * v;
  __syncthreads();
  for (int s = 128; s > 0; s >>= 1) {
    if (d < s) red[d] += red[d + s];
    __syncthreads();
  }
  dst[j * 256 + d] = v / sqrtf(red[0]);
}

// ---------------- host ----------------
extern "C" void kernel_launch(void* const* d_in, const int* in_sizes, int n_in,
                              void* d_out, int out_size, void* d_ws, size_t ws_size,
                              hipStream_t stream) {
  (void)in_sizes; (void)n_in; (void)out_size; (void)ws_size;
  const float* input   = (const float*)d_in[0];
  const float* objects = (const float*)d_in[1];
  const float* Wc  = (const float*)d_in[3];
  const float* bc  = (const float*)d_in[4];
  const float* Wr  = (const float*)d_in[5];
  const float* br  = (const float*)d_in[6];
  const float* Wof = (const float*)d_in[7];
  const float* bof = (const float*)d_in[8];
  const float* Wrf = (const float*)d_in[9];
  const float* brf = (const float*)d_in[10];
  const float* Wofc= (const float*)d_in[11];
  const float* bofc= (const float*)d_in[12];
  const float* Wrfc= (const float*)d_in[13];
  const float* brfc= (const float*)d_in[14];
  float* out = (float*)d_out;
  float* ws  = (float*)d_ws;

  size_t off = 0;
  auto allocf = [&](size_t nf) { float* p = ws + off; off += (nf + 3) & ~size_t(3); return p; };
  auto allocb = [&](size_t ne) { return (__bf16*)allocf((ne + 1) / 2); };

  float*  cat    = allocf((size_t)640 * 8192);        // [640][8192] f32 (ctx 0..255, rel 256..639)
  float*  bcat   = allocf(640);
  __bf16* wA_cat = allocb((size_t)640 * 256);
  __bf16* wA_of  = allocb((size_t)256 * 512);
  __bf16* wA_rf  = allocb((size_t)256 * 896);
  __bf16* wA_rfc = allocb((size_t)256 * 12544);
  __bf16* wA_ofc = allocb((size_t)256 * 12544);
  __bf16* BT_in  = allocb((size_t)8192 * 256);
  float*  wx_o   = allocf(128 * 224);
  float*  wy_o   = allocf(128 * 224);
  float*  ia_o   = allocf(128);
  float*  imgbox = allocf(4);
  float*  wx_i   = allocf(224);
  float*  wy_i   = allocf(224);
  float*  ia_i   = allocf(4);
  float*  uni    = allocf(2048 * 4);
  float*  wx_u   = allocf(2048 * 224);
  float*  wy_u   = allocf(2048 * 224);
  float*  ia_u   = allocf(2048);
  float*  bimap  = allocf(128 * 49);
  float*  simap  = allocf(2048 * 49);
  float*  oimap  = allocf(2048 * 49);
  // per-image scratch (stream-serialized reuse)
  float*  ctxp   = allocf(256 * 49);
  float*  objp   = allocf(16 * 256 * 49);
  float*  objf   = allocf((size_t)256 * 784);
  float*  relp   = allocf((size_t)256 * 384 * 49);
  float*  relf   = allocf((size_t)256 * 12544);
  float*  o_tmp  = allocf(256 * 16);
  float*  r_tmp  = allocf(256 * 256);
  __bf16* BT_oin = allocb((size_t)784 * 512);
  __bf16* BT_rin = allocb((size_t)12544 * 896);
  __bf16* BT_rf  = allocb((size_t)256 * 12544);
  __bf16* BT_of  = allocb((size_t)16 * 12544);

  const int TB = 256;
  auto blk = [](long long n, int tb) { return (int)((n + tb - 1) / tb); };

  // ---- setup ----
  k_init_imgbox<<<1, 1, 0, stream>>>(imgbox);
  k_roi_weights<<<blk(128 * 224, TB), TB, 0, stream>>>(objects, 128, wx_o, wy_o, ia_o);
  k_roi_weights<<<1, TB, 0, stream>>>(imgbox, 1, wx_i, wy_i, ia_i);
  k_unions<<<blk(2048, TB), TB, 0, stream>>>(objects, uni);
  k_roi_weights<<<blk(2048 * 224, TB), TB, 0, stream>>>(uni, 2048, wx_u, wy_u, ia_u);
  k_imap_img<<<blk(128 * 49, TB), TB, 0, stream>>>(objects, bimap);
  k_imap_pairs<<<blk(2048 * 49, TB), TB, 0, stream>>>(objects, uni, simap, oimap);

  // weight conversion (once)
  k_cvt_bf16<<<blk(256 * 256, TB), TB, 0, stream>>>(Wc, wA_cat, 256 * 256);
  k_cvt_bf16<<<blk(384 * 256, TB), TB, 0, stream>>>(Wr, wA_cat + 256 * 256, 384 * 256);
  k_cvt_bf16<<<blk(256 * 512, TB), TB, 0, stream>>>(Wof, wA_of, 256 * 512);
  k_cvt_bf16<<<blk(256 * 896, TB), TB, 0, stream>>>(Wrf, wA_rf, 256 * 896);
  k_cvt_bf16<<<blk((long long)256 * 12544, TB), TB, 0, stream>>>(Wrfc, wA_rfc, 256 * 12544);
  k_cvt_bf16<<<blk((long long)256 * 12544, TB), TB, 0, stream>>>(Wofc, wA_ofc, 256 * 12544);
  hipMemcpyAsync(bcat, bc, 256 * sizeof(float), hipMemcpyDeviceToDevice, stream);
  hipMemcpyAsync(bcat + 256, br, 384 * sizeof(float), hipMemcpyDeviceToDevice, stream);

  // G1: fused ctx+rel 1x1 conv. M=640 (Mt=40), N=8192, K=256, NW=2 -> 10240 waves
  k_pack_inT<<<blk((long long)8192 * 256, TB), TB, 0, stream>>>(input, BT_in);
  k_gemm_bf16<2><<<(40 * 256) * 32 / TB, TB, 0, stream>>>(wA_cat, BT_in, bcat, cat, 40, 8192, 256);

  for (int b = 0; b < 8; ++b) {
    const float* ctx_b = cat + b * 1024;                     // cstride 8192
    const float* rel_b = cat + (size_t)256 * 8192 + b * 1024;
    const float* in_b  = input + (size_t)b * 262144;         // cstride 1024

    k_prroi_pool<<<blk(1 * 256 * 7, TB), TB, 0, stream>>>(ctx_b, 256, 8192, wy_i, wx_i, ia_i, 1, ctxp);
    k_prroi_pool<<<blk(16 * 256 * 7, TB), TB, 0, stream>>>(in_b, 256, 1024, wy_o + b * 16 * 224,
                                                           wx_o + b * 16 * 224, ia_o + b * 16, 16, objp);

    // G2: obj_feat. M=256 (Mt=16), N=784 (49 tiles, odd) -> NW=1, K=512 -> 784 waves
    k_pack_objinT<<<blk((long long)784 * 512, TB), TB, 0, stream>>>(objp, ctxp, bimap + b * 784, BT_oin);
    k_gemm_bf16<1><<<(16 * 49) * 32 / TB, TB, 0, stream>>>(wA_of, BT_oin, bof, objf, 16, 784, 512);

    k_prroi_pool<<<blk(256 * 384 * 7, TB), TB, 0, stream>>>(rel_b, 384, 8192, wy_u + b * 256 * 224,
                                                            wx_u + b * 256 * 224, ia_u + b * 256, 256, relp);

    // G3: rel_feat. M=256 (Mt=16), N=12544 (784 tiles) NW=2 -> 6272 waves, K=896
    k_pack_relinT<<<blk((long long)12544 * 896, TB), TB, 0, stream>>>(objf, relp, simap + b * 12544,
                                                                      oimap + b * 12544, BT_rin);
    k_gemm_bf16<2><<<(16 * 392) * 32 / TB, TB, 0, stream>>>(wA_rf, BT_rin, brf, relf, 16, 12544, 896);

    // G4: r FC. M=256, N=256, K=12544, NW=2 -> 128 waves
    k_pack_relfT<<<blk((long long)256 * 12544, TB), TB, 0, stream>>>(relf, BT_rf);
    k_gemm_bf16<2><<<(16 * 8) * 32 / TB, TB, 0, stream>>>(wA_rfc, BT_rf, brfc, r_tmp, 16, 256, 12544);

    // G5: o FC. M=256, N=16, K=12544, NW=1 -> 16 waves
    k_pack_objfT<<<blk((long long)16 * 12544, TB), TB, 0, stream>>>(objf, BT_of);
    k_gemm_bf16<1><<<(16 * 1) * 32 / TB, TB, 0, stream>>>(wA_ofc, BT_of, bofc, o_tmp, 16, 16, 12544);

    k_normalize<<<16, 256, 0, stream>>>(o_tmp, 16, out + (size_t)b * 16 * 256);
    k_normalize<<<256, 256, 0, stream>>>(r_tmp, 256, out + 32768 + (size_t)b * 65536);
  }
}
